// HeteroGraphSAGE_LINK_5033701670915
// MI455X (gfx1250) — compile-verified, matrix operand order checked by reading, not run on hardware
//
#include <hip/hip_runtime.h>

// ---------------- problem constants ----------------
static constexpr int kNU   = 100000;
static constexpr int kNI   = 100000;
static constexpr int kNH   = 200000;   // NU + NI
static constexpr int kC    = 128;
static constexpr int kPD   = 64;
static constexpr int kPH   = 128;
static constexpr int kEHET = 500000;
static constexpr int kEHOM = 1000000;

// ---------------- WMMA types -----------------------
typedef __attribute__((ext_vector_type(16))) __bf16          v16bf;
typedef __attribute__((ext_vector_type(2)))  __bf16          v2bf;
typedef __attribute__((ext_vector_type(8)))  float           v8f;

#define WMMA_BF16(a, b, c) \
  __builtin_amdgcn_wmma_f32_16x16x32_bf16(false, (a), false, (b), (short)0, (c), false, false)

__device__ __forceinline__ unsigned short f2bf(float f) {
  unsigned u = __float_as_uint(f);
  unsigned r = u + 0x7FFFu + ((u >> 16) & 1u);   // round-to-nearest-even
  return (unsigned short)(r >> 16);
}

// Pack two f32 into two bf16 (one dword). Prefer HW cvt; otherwise a single
// v_perm_b32 concatenating the high halves: D = {y[31:16], x[31:16]}.
__device__ __forceinline__ unsigned pk2(float x, float y) {
#if __has_builtin(__builtin_amdgcn_cvt_pk_bf16_f32)
  v2bf r = __builtin_amdgcn_cvt_pk_bf16_f32(x, y);
  return __builtin_bit_cast(unsigned, r);
#else
  return __builtin_amdgcn_perm(__float_as_uint(y), __float_as_uint(x), 0x07060302u);
#endif
}

struct Pk8 { unsigned w[8]; };

// A-fragment (16x32 bf16, M x K) from a row-major f32 row pointer at kbase.
// Lane layout (ISA 7.12.2): lane = m + 16*hi; per VGPR v, k = 16*(v/4) + 8*hi + 2*(v%4) + half.
// The 4 pairs of a lane-group are 8 contiguous floats -> two b128 loads + 4 perms per group.
__device__ __forceinline__ v16bf make_a(const float* __restrict__ row, int hi) {
  Pk8 u;
#pragma unroll
  for (int g = 0; g < 2; ++g) {
    const float* p = row + 16 * g + 8 * hi;
    float4 f0 = *(const float4*)p;
    float4 f1 = *(const float4*)(p + 4);
    u.w[4 * g + 0] = pk2(f0.x, f0.y);
    u.w[4 * g + 1] = pk2(f0.z, f0.w);
    u.w[4 * g + 2] = pk2(f1.x, f1.y);
    u.w[4 * g + 3] = pk2(f1.z, f1.w);
  }
  return __builtin_bit_cast(v16bf, u);
}

// B-fragment from repacked weights: frag = tileN*Kt + tileK, 512 ushorts/frag.
__device__ __forceinline__ v16bf load_b(const unsigned short* __restrict__ wb,
                                        int frag, int lane) {
  return *(const v16bf*)(wb + (size_t)frag * 512 + (size_t)lane * 16);
}

// ---------------- utility kernels ------------------
__global__ void k_zero(float* __restrict__ p, long long n4) {
  long long i = (long long)blockIdx.x * blockDim.x + threadIdx.x;
  if (i < n4) ((float4*)p)[i] = make_float4(0.f, 0.f, 0.f, 0.f);
}

// Repack a row-major (K,N) f32 weight into WMMA B-fragment order (bf16, RNE).
__global__ void k_repack(const float* __restrict__ src, unsigned short* __restrict__ dst,
                         int K, int N) {
  int gid = blockIdx.x * blockDim.x + threadIdx.x;
  int Kt = K >> 5;
  int total = Kt * (N >> 4) * 32;
  if (gid >= total) return;
  int lane = gid & 31;
  int f = gid >> 5;
  int tileK = f % Kt, tileN = f / Kt;
  int nloc = lane & 15, hi = lane >> 4;
  int n = tileN * 16 + nloc;
  unsigned short* o = dst + (size_t)f * 512 + (size_t)lane * 16;
#pragma unroll
  for (int v = 0; v < 8; ++v) {
#pragma unroll
    for (int hf = 0; hf < 2; ++hf) {
      int kl = 16 * (v >> 2) + 8 * hi + 2 * (v & 3) + hf;
      o[v * 2 + hf] = f2bf(src[(size_t)(tileK * 32 + kl) * N + n]);
    }
  }
}

// ---------------- GIN aggregation (layer-invariant) -------------
__global__ void k_gin_scatter(const float* __restrict__ PE, const int* __restrict__ edge,
                              float* __restrict__ agg) {
  long long gid = (long long)blockIdx.x * blockDim.x + threadIdx.x;
  int e = (int)(gid >> 4);
  if (e >= kEHOM) return;
  int q = (int)(gid & 15) * 4;
  int s = edge[e];
  int d = edge[kEHOM + e];
  float4 v = *(const float4*)(PE + (size_t)s * kPD + q);
  float* o = agg + (size_t)d * kPD + q;
  atomicAdd(o + 0, v.x); atomicAdd(o + 1, v.y);
  atomicAdd(o + 2, v.z); atomicAdd(o + 3, v.w);
}

// ---------------- GIN MLP1: h = relu(((1+eps)*PE + agg) @ W1 + b1) ----
// Block = 32 rows x 128 cols, 8 waves: rowgroup = wv>>2, N-tile pair = (wv&3)*2.
__global__ void __launch_bounds__(256)
k_gin_mlp1(const float* __restrict__ PE, const float* __restrict__ agg,
           const unsigned short* __restrict__ wb, const float* __restrict__ bias1,
           const float* __restrict__ phi_eps, int l, float* __restrict__ h) {
  int wv = threadIdx.x >> 5, lane = threadIdx.x & 31;
  int rg = wv >> 2, nb = (wv & 3) * 2;
  int m = lane & 15, hi = lane >> 4;
  size_t r0 = (size_t)blockIdx.x * 32 + rg * 16;
  size_t row = r0 + m;
  float e1 = 1.0f + phi_eps[l];
  v8f c00 = {}, c01 = {}, c10 = {}, c11 = {};
#pragma unroll
  for (int kt = 0; kt < 2; ++kt) {
    const float* pr = PE  + row * kPD + kt * 32;
    const float* ar = agg + row * kPD + kt * 32;
    Pk8 u;
#pragma unroll
    for (int g = 0; g < 2; ++g) {
      const float* p = pr + 16 * g + 8 * hi;
      const float* q = ar + 16 * g + 8 * hi;
      float4 f0 = *(const float4*)p, f1 = *(const float4*)(p + 4);
      float4 g0 = *(const float4*)q, g1 = *(const float4*)(q + 4);
      u.w[4 * g + 0] = pk2(fmaf(e1, f0.x, g0.x), fmaf(e1, f0.y, g0.y));
      u.w[4 * g + 1] = pk2(fmaf(e1, f0.z, g0.z), fmaf(e1, f0.w, g0.w));
      u.w[4 * g + 2] = pk2(fmaf(e1, f1.x, g1.x), fmaf(e1, f1.y, g1.y));
      u.w[4 * g + 3] = pk2(fmaf(e1, f1.z, g1.z), fmaf(e1, f1.w, g1.w));
    }
    v16bf a = __builtin_bit_cast(v16bf, u);
    v16bf bb0 = load_b(wb, (nb + 0) * 2 + kt, lane);
    v16bf bb1 = load_b(wb, (nb + 1) * 2 + kt, lane);
    if (kt == 0) { c00 = WMMA_BF16(a, bb0, c00); c10 = WMMA_BF16(a, bb1, c10); }
    else         { c01 = WMMA_BF16(a, bb0, c01); c11 = WMMA_BF16(a, bb1, c11); }
  }
  int col0 = nb * 16 + m, col1 = col0 + 16;
  float bs0 = bias1[col0], bs1 = bias1[col1];
#pragma unroll
  for (int v = 0; v < 8; ++v) {
    size_t r = r0 + v + 8 * hi;
    h[r * kPH + col0] = fmaxf(c00[v] + c01[v] + bs0, 0.f);
    h[r * kPH + col1] = fmaxf(c10[v] + c11[v] + bs1, 0.f);
  }
}

// ---------------- GIN MLP2: pe_l = h @ W2 + b2 ----------------------
// Block = 64 rows x 64 cols, 8 waves: rowgroup = wv>>1, N-tile pair = (wv&1)*2.
__global__ void __launch_bounds__(256)
k_gin_mlp2(const float* __restrict__ h, const unsigned short* __restrict__ wb,
           const float* __restrict__ bias2, float* __restrict__ pel) {
  int wv = threadIdx.x >> 5, lane = threadIdx.x & 31;
  int rg = wv >> 1, nb = (wv & 1) * 2;
  int m = lane & 15, hi = lane >> 4;
  size_t r0 = (size_t)blockIdx.x * 64 + rg * 16;
  const float* arow = h + (r0 + m) * kPH;
  v8f c0a = {}, c0b = {}, c1a = {}, c1b = {};
#pragma unroll
  for (int kt = 0; kt < 4; ++kt) {
    v16bf a = make_a(arow + kt * 32, hi);
    v16bf bb0 = load_b(wb, (nb + 0) * 4 + kt, lane);
    v16bf bb1 = load_b(wb, (nb + 1) * 4 + kt, lane);
    if (kt & 1) { c0b = WMMA_BF16(a, bb0, c0b); c1b = WMMA_BF16(a, bb1, c1b); }
    else        { c0a = WMMA_BF16(a, bb0, c0a); c1a = WMMA_BF16(a, bb1, c1a); }
  }
  int col0 = nb * 16 + m, col1 = col0 + 16;
  float bs0 = bias2[col0], bs1 = bias2[col1];
#pragma unroll
  for (int v = 0; v < 8; ++v) {
    size_t r = r0 + v + 8 * hi;
    pel[r * kPD + col0] = c0a[v] + c0b[v] + bs0;
    pel[r * kPD + col1] = c1a[v] + c1b[v] + bs1;
  }
}

// ---------------- x_new = [x | pe_l] @ pe_W + pe_b  (K=192, 6 wmma/tile) ----
__global__ void __launch_bounds__(256)
k_xproj(const float* __restrict__ xu, const float* __restrict__ xi,
        const float* __restrict__ pel, const unsigned short* __restrict__ wb,
        const float* __restrict__ pbias, float* __restrict__ xnew) {
  int wv = threadIdx.x >> 5, lane = threadIdx.x & 31;
  int rg = wv >> 2, nb = (wv & 3) * 2;
  int m = lane & 15, hi = lane >> 4;
  size_t r0 = (size_t)blockIdx.x * 32 + rg * 16;
  size_t row = r0 + m;
  const float* xrow = (row < (size_t)kNU) ? (xu + row * kC) : (xi + (row - kNU) * kC);
  const float* prow = pel + row * kPD;
  v8f c0a = {}, c0b = {}, c1a = {}, c1b = {};
#pragma unroll
  for (int kt = 0; kt < 6; ++kt) {
    const float* src = (kt < 4) ? (xrow + kt * 32) : (prow + (kt - 4) * 32);
    v16bf a = make_a(src, hi);
    v16bf bb0 = load_b(wb, (nb + 0) * 6 + kt, lane);
    v16bf bb1 = load_b(wb, (nb + 1) * 6 + kt, lane);
    if (kt & 1) { c0b = WMMA_BF16(a, bb0, c0b); c1b = WMMA_BF16(a, bb1, c1b); }
    else        { c0a = WMMA_BF16(a, bb0, c0a); c1a = WMMA_BF16(a, bb1, c1a); }
  }
  int col0 = nb * 16 + m, col1 = col0 + 16;
  float bs0 = pbias[col0], bs1 = pbias[col1];
#pragma unroll
  for (int v = 0; v < 8; ++v) {
    size_t r = r0 + v + 8 * hi;
    xnew[r * kC + col0] = c0a[v] + c0b[v] + bs0;
    xnew[r * kC + col1] = c1a[v] + c1b[v] + bs1;
  }
}

// ---------------- SAGE neighbor scatter: msum[dst]+=x[src], cnt[dst]+=1 ----
__global__ void k_sage_scatter(const float* __restrict__ x, const int* __restrict__ edge,
                               int srcBase, int dstBase,
                               float* __restrict__ msum, float* __restrict__ cnt) {
  long long gid = (long long)blockIdx.x * blockDim.x + threadIdx.x;
  int e = (int)(gid >> 5);
  if (e >= kEHET) return;
  int q = (int)(gid & 31) * 4;
  int s = edge[e] + srcBase;
  int d = edge[kEHET + e] + dstBase;
  float4 v = *(const float4*)(x + (size_t)s * kC + q);
  float* o = msum + (size_t)d * kC + q;
  atomicAdd(o + 0, v.x); atomicAdd(o + 1, v.y);
  atomicAdd(o + 2, v.z); atomicAdd(o + 3, v.w);
  if ((gid & 31) == 0) atomicAdd(&cnt[d], 1.0f);
}

// -------- SAGE: out = relu(LN(mean @ Wl + bl + x @ Wr) * g + b)  -----------
// Mean scale is per-row of A, so fold it into the epilogue:
//   (inv*msum) @ Wl == inv * (msum @ Wl).
// Block = 32 rows x 128 cols, 8 waves; 4 independent WMMA chains interleaved.
__global__ void __launch_bounds__(256)
k_sage(const float* __restrict__ xnew, const float* __restrict__ msum,
       const float* __restrict__ cnt,
       const unsigned short* __restrict__ wl_u, const unsigned short* __restrict__ wr_u,
       const unsigned short* __restrict__ wl_i, const unsigned short* __restrict__ wr_i,
       const float* __restrict__ bl_u, const float* __restrict__ bl_i,
       const float* __restrict__ g_u, const float* __restrict__ bb_u,
       const float* __restrict__ g_i, const float* __restrict__ bb_i,
       float* __restrict__ xout) {
  __shared__ float tile[32][kC + 1];
  int wv = threadIdx.x >> 5, lane = threadIdx.x & 31;
  int rg = wv >> 2, nb = (wv & 3) * 2;
  int m = lane & 15, hi = lane >> 4;
  size_t r0 = (size_t)blockIdx.x * 32;
  bool isU = (r0 < (size_t)kNU);
  const unsigned short* wl = isU ? wl_u : wl_i;
  const unsigned short* wr = isU ? wr_u : wr_i;
  const float* bl = isU ? bl_u : bl_i;
  const float* g  = isU ? g_u : g_i;
  const float* bb = isU ? bb_u : bb_i;

  size_t row = r0 + rg * 16 + m;
  const float* mrow = msum + row * kC;
  const float* xrow = xnew + row * kC;

  v8f cm0 = {}, cm1 = {}, cx0 = {}, cx1 = {};
#pragma unroll
  for (int kt = 0; kt < 4; ++kt) {
    v16bf am = make_a(mrow + kt * 32, hi);
    v16bf ax = make_a(xrow + kt * 32, hi);
    v16bf bl0 = load_b(wl, (nb + 0) * 4 + kt, lane);
    v16bf bl1 = load_b(wl, (nb + 1) * 4 + kt, lane);
    v16bf br0 = load_b(wr, (nb + 0) * 4 + kt, lane);
    v16bf br1 = load_b(wr, (nb + 1) * 4 + kt, lane);
    cm0 = WMMA_BF16(am, bl0, cm0);
    cx0 = WMMA_BF16(ax, br0, cx0);
    cm1 = WMMA_BF16(am, bl1, cm1);
    cx1 = WMMA_BF16(ax, br1, cx1);
  }

  int col0 = nb * 16 + m, col1 = col0 + 16;
  float bs0 = bl[col0], bs1 = bl[col1];
#pragma unroll
  for (int v = 0; v < 8; ++v) {
    int rl = rg * 16 + v + 8 * hi;
    float cn = cnt[r0 + rl];
    float iv = 1.0f / (cn > 1.0f ? cn : 1.0f);
    tile[rl][col0] = fmaf(cm0[v], iv, cx0[v]) + bs0;
    tile[rl][col1] = fmaf(cm1[v], iv, cx1[v]) + bs1;
  }
  __syncthreads();

  if (threadIdx.x < 32) {
    int rl = (int)threadIdx.x;
    float s = 0.f;
#pragma unroll 4
    for (int cc = 0; cc < kC; ++cc) s += tile[rl][cc];
    float mu = s * (1.0f / kC);
    float vs = 0.f;
#pragma unroll 4
    for (int cc = 0; cc < kC; ++cc) { float d = tile[rl][cc] - mu; vs += d * d; }
    float rstd = rsqrtf(vs * (1.0f / kC) + 1e-5f);
    size_t orow = (r0 + rl) * kC;
#pragma unroll 4
    for (int cc = 0; cc < kC; ++cc) {
      float y = (tile[rl][cc] - mu) * rstd * g[cc] + bb[cc];
      xout[orow + cc] = y > 0.f ? y : 0.f;
    }
  }
}

// ---------------- host orchestration ----------------
extern "C" void kernel_launch(void* const* d_in, const int* in_sizes, int n_in,
                              void* d_out, int out_size, void* d_ws, size_t ws_size,
                              hipStream_t stream) {
  const float* x_user  = (const float*)d_in[0];
  const float* x_item  = (const float*)d_in[1];
  const float* PE      = (const float*)d_in[2];
  const float* ui_Wl   = (const float*)d_in[3];
  const float* ui_bl   = (const float*)d_in[4];
  const float* ui_Wr   = (const float*)d_in[5];
  const float* iu_Wl   = (const float*)d_in[6];
  const float* iu_bl   = (const float*)d_in[7];
  const float* iu_Wr   = (const float*)d_in[8];
  const float* ln_ug   = (const float*)d_in[9];
  const float* ln_ub   = (const float*)d_in[10];
  const float* ln_ig   = (const float*)d_in[11];
  const float* ln_ib   = (const float*)d_in[12];
  const float* phi_eps = (const float*)d_in[13];
  const float* phi_W1  = (const float*)d_in[14];
  const float* phi_b1  = (const float*)d_in[15];
  const float* phi_W2  = (const float*)d_in[16];
  const float* phi_b2  = (const float*)d_in[17];
  const float* pe_W    = (const float*)d_in[18];
  const float* pe_b    = (const float*)d_in[19];
  const int*   e_ui    = (const int*)d_in[20];
  const int*   e_iu    = (const int*)d_in[21];
  const int*   e_hom   = (const int*)d_in[22];
  (void)in_sizes; (void)n_in; (void)out_size; (void)ws_size;

  // workspace layout (floats); hbuf and msum share storage (disjoint lifetimes)
  float* ws = (float*)d_ws;
  size_t off = 0;
  float* agg  = ws + off; off += (size_t)kNH * kPD;
  float* hbuf = ws + off; off += (size_t)kNH * kPH;
  float* pel  = ws + off; off += (size_t)kNH * kPD;
  float* xnew = ws + off; off += (size_t)kNH * kC;
  float* xcur = ws + off; off += (size_t)kNH * kC;
  float* cnt  = ws + off; off += (size_t)kNH;
  float* msum = hbuf;
  unsigned short* wb = (unsigned short*)(ws + off);

  const size_t F_W1 = 64 * 128, F_W2 = 128 * 64, F_PEW = 192 * 128, F_SQ = 128 * 128;
  const size_t LWB = F_W1 + F_W2 + F_PEW + 4 * F_SQ;

  // ---- one-time weight repack (bf16, fragment order) ----
  for (int l = 0; l < 2; ++l) {
    unsigned short* w = wb + (size_t)l * LWB;
    auto rp = [&](const float* s, unsigned short* d, int K, int N) {
      int threads = (K / 32) * (N / 16) * 32;
      k_repack<<<(threads + 255) / 256, 256, 0, stream>>>(s, d, K, N);
    };
    rp(phi_W1 + (size_t)l * 64 * 128,  w,                 64, 128);
    rp(phi_W2 + (size_t)l * 128 * 64,  w + F_W1,         128,  64);
    rp(pe_W   + (size_t)l * 192 * 128, w + F_W1 + F_W2,  192, 128);
    unsigned short* wsage = w + F_W1 + F_W2 + F_PEW;
    rp(ui_Wl + (size_t)l * 128 * 128, wsage,             128, 128);
    rp(ui_Wr + (size_t)l * 128 * 128, wsage + F_SQ,      128, 128);
    rp(iu_Wl + (size_t)l * 128 * 128, wsage + 2 * F_SQ,  128, 128);
    rp(iu_Wr + (size_t)l * 128 * 128, wsage + 3 * F_SQ,  128, 128);
  }

  // ---- GIN segment-sum over hom edges (PE is constant across layers) ----
  {
    long long n4 = (long long)kNH * kPD / 4;
    k_zero<<<(int)((n4 + 255) / 256), 256, 0, stream>>>(agg, n4);
    k_gin_scatter<<<(kEHOM * 16) / 256, 256, 0, stream>>>(PE, e_hom, agg);
  }

  for (int l = 0; l < 2; ++l) {
    unsigned short* w = wb + (size_t)l * LWB;
    unsigned short* wsage = w + F_W1 + F_W2 + F_PEW;

    k_gin_mlp1<<<kNH / 32, 256, 0, stream>>>(PE, agg, w, phi_b1 + (size_t)l * kPH,
                                             phi_eps, l, hbuf);
    k_gin_mlp2<<<kNH / 64, 256, 0, stream>>>(hbuf, w + F_W1, phi_b2 + (size_t)l * kPD, pel);

    const float* xu = (l == 0) ? x_user : xcur;
    const float* xi = (l == 0) ? x_item : xcur + (size_t)kNU * kC;
    k_xproj<<<kNH / 32, 256, 0, stream>>>(xu, xi, pel, w + F_W1 + F_W2,
                                          pe_b + (size_t)l * kC, xnew);

    long long n4m = (long long)kNH * kC / 4;
    k_zero<<<(int)((n4m + 255) / 256), 256, 0, stream>>>(msum, n4m);
    long long n4c = (long long)kNH / 4;
    k_zero<<<(int)((n4c + 255) / 256), 256, 0, stream>>>(cnt, n4c);

    k_sage_scatter<<<(kEHET * 32) / 256, 256, 0, stream>>>(xnew, e_ui, 0, kNU, msum, cnt);
    k_sage_scatter<<<(kEHET * 32) / 256, 256, 0, stream>>>(xnew, e_iu, kNU, 0, msum, cnt);

    float* xout = (l == 1) ? (float*)d_out : xcur;
    k_sage<<<kNH / 32, 256, 0, stream>>>(
        xnew, msum, cnt,
        /*user (iu) */ wsage + 2 * F_SQ, wsage + 3 * F_SQ,
        /*item (ui) */ wsage,            wsage + F_SQ,
        iu_bl + (size_t)l * kC, ui_bl + (size_t)l * kC,
        ln_ug + (size_t)l * kC, ln_ub + (size_t)l * kC,
        ln_ig + (size_t)l * kC, ln_ib + (size_t)l * kC,
        xout);
  }
}
